// EdgeMLP_12429635354687
// MI455X (gfx1250) — compile-verified, matrix operand order checked by reading, not run on hardware
//
#include <hip/hip_runtime.h>

typedef __attribute__((ext_vector_type(16))) __bf16 v16bf;
typedef __attribute__((ext_vector_type(8)))  float  v8f;

#define WAVES_PER_BLOCK 8
#define TPB (WAVES_PER_BLOCK * 32)

union Frag {
    v16bf  v;
    uint4  q[2];
    __bf16 e[16];
};

__global__ __launch_bounds__(TPB, 1)
void edge_mlp_wmma_kernel(const float* __restrict__ srcF,   // (E,128)
                          const float* __restrict__ edgeF,  // (E,64)
                          const float* __restrict__ W1,     // (192,128)
                          const float* __restrict__ B1,     // (128)
                          const float* __restrict__ GAMMA,  // (128)
                          const float* __restrict__ BETA,   // (128)
                          const float* __restrict__ W2,     // (128,128)
                          const float* __restrict__ B2,     // (128)
                          float* __restrict__ out,          // (E,128)
                          int E, int nTiles)
{
    // Weights transposed to [n][k] in bf16 so B-fragment lanes read contiguous 16B.
    __shared__ __align__(16) __bf16 sW1[128][192];                 // 48 KB
    __shared__ __align__(16) __bf16 sW2[128][128];                 // 32 KB
    __shared__ __align__(16) __bf16 sH[WAVES_PER_BLOCK][16][128];  // 32 KB

    const int tid = threadIdx.x;

    // ---- stage weights (fp32 -> bf16, transpose) ----
    for (int i = tid; i < 192 * 128; i += TPB) {
        int k = i >> 7, n = i & 127;
        sW1[n][k] = (__bf16)W1[i];
    }
    for (int i = tid; i < 128 * 128; i += TPB) {
        int k = i >> 7, n = i & 127;
        sW2[n][k] = (__bf16)W2[i];
    }
    __syncthreads();

    const int lane = tid & 31;
    const int wave = tid >> 5;
    const int c    = lane & 15;   // M for A-frags / N for B,C,D frags
    const int half = lane >> 4;   // K-chunk select within fragments

    // per-lane column constants (column = 16*t + c)
    float g[8], bt[8], bb1[8], bb2[8];
#pragma unroll
    for (int t = 0; t < 8; ++t) {
        int col = 16 * t + c;
        g[t]   = GAMMA[col];
        bt[t]  = BETA[col];
        bb1[t] = B1[col];
        bb2[t] = B2[col];
    }

    const int waveId = blockIdx.x * WAVES_PER_BLOCK + wave;
    const int nWaves = gridDim.x * WAVES_PER_BLOCK;
    const v8f vzero = {0.f, 0.f, 0.f, 0.f, 0.f, 0.f, 0.f, 0.f};

    for (int tile = waveId; tile < nTiles; tile += nWaves) {
        int row = tile * 16 + c;          // edge row this lane loads for A-frags
        if (row >= E) row = E - 1;        // clamp tail (stores are guarded below)

        v8f acc[8];
#pragma unroll
        for (int t = 0; t < 8; ++t) acc[t] = vzero;

        // ---- GEMM1: h = [src|edge](16x192) @ W1(192x128), K in 6 steps of 32 ----
#pragma unroll
        for (int ks = 0; ks < 6; ++ks) {
            const int kw = ks * 32 + 8 * half;   // first K chunk of this fragment
            const float* rp = (ks < 4)
                ? (srcF  + (size_t)row * 128 + kw)
                : (edgeF + (size_t)row * 64  + (kw - 128));
            float4 a0 = *(const float4*)(rp);
            float4 a1 = *(const float4*)(rp + 4);
            float4 a2 = *(const float4*)(rp + 16);   // second chunk: K + 16
            float4 a3 = *(const float4*)(rp + 20);
            Frag fa;
            fa.e[0]=(__bf16)a0.x;  fa.e[1]=(__bf16)a0.y;  fa.e[2]=(__bf16)a0.z;  fa.e[3]=(__bf16)a0.w;
            fa.e[4]=(__bf16)a1.x;  fa.e[5]=(__bf16)a1.y;  fa.e[6]=(__bf16)a1.z;  fa.e[7]=(__bf16)a1.w;
            fa.e[8]=(__bf16)a2.x;  fa.e[9]=(__bf16)a2.y;  fa.e[10]=(__bf16)a2.z; fa.e[11]=(__bf16)a2.w;
            fa.e[12]=(__bf16)a3.x; fa.e[13]=(__bf16)a3.y; fa.e[14]=(__bf16)a3.z; fa.e[15]=(__bf16)a3.w;
#pragma unroll
            for (int t = 0; t < 8; ++t) {
                Frag fb;
                const __bf16* wp = &sW1[16 * t + c][ks * 32 + 8 * half];
                fb.q[0] = *(const uint4*)(wp);
                fb.q[1] = *(const uint4*)(wp + 16);
                acc[t] = __builtin_amdgcn_wmma_f32_16x16x32_bf16(
                    false, fa.v, false, fb.v, (short)0, acc[t], false, false);
            }
        }

        // ---- bias + LayerNorm + SiLU; write bf16 h-tile to per-wave LDS ----
#pragma unroll
        for (int m = 0; m < 8; ++m) {
            float sm = 0.f, sq = 0.f;
#pragma unroll
            for (int t = 0; t < 8; ++t) {
                float v = acc[t][m] + bb1[t];
                acc[t][m] = v;
                sm += v; sq += v * v;
            }
            // reduce across the 16-lane half (xor<16 stays within each half)
#pragma unroll
            for (int off = 1; off < 16; off <<= 1) {
                sm += __shfl_xor(sm, off, 32);
                sq += __shfl_xor(sq, off, 32);
            }
            float mu   = sm * (1.0f / 128.0f);
            float var  = sq * (1.0f / 128.0f) - mu * mu;
            float rstd = rsqrtf(var + 1e-5f);
            int rrow = m + 8 * half;   // C-layout row for this VGPR/half
#pragma unroll
            for (int t = 0; t < 8; ++t) {
                float v = (acc[t][m] - mu) * rstd * g[t] + bt[t];
                v = v / (1.0f + __expf(-v));          // SiLU
                sH[wave][rrow][16 * t + c] = (__bf16)v;
            }
        }
        // same-wave LDS ops are in-order: no barrier needed before re-reading sH

        // ---- GEMM2: y = h(16x128) @ W2(128x128), K in 4 steps of 32 ----
        v8f acc2[8];
#pragma unroll
        for (int t = 0; t < 8; ++t) acc2[t] = vzero;
#pragma unroll
        for (int ks = 0; ks < 4; ++ks) {
            Frag fa;
            const __bf16* hp = &sH[wave][c][ks * 32 + 8 * half];
            fa.q[0] = *(const uint4*)(hp);
            fa.q[1] = *(const uint4*)(hp + 16);
#pragma unroll
            for (int t = 0; t < 8; ++t) {
                Frag fb;
                const __bf16* wp = &sW2[16 * t + c][ks * 32 + 8 * half];
                fb.q[0] = *(const uint4*)(wp);
                fb.q[1] = *(const uint4*)(wp + 16);
                acc2[t] = __builtin_amdgcn_wmma_f32_16x16x32_bf16(
                    false, fa.v, false, fb.v, (short)0, acc2[t], false, false);
            }
        }

        // ---- bias + SiLU, store f32 output ----
#pragma unroll
        for (int m = 0; m < 8; ++m) {
            int orow = tile * 16 + m + 8 * half;
            bool ok = orow < E;
#pragma unroll
            for (int t = 0; t < 8; ++t) {
                float v = acc2[t][m] + bb2[t];
                v = v / (1.0f + __expf(-v));
                if (ok) out[(size_t)orow * 128 + 16 * t + c] = v;
            }
        }
    }
}

extern "C" void kernel_launch(void* const* d_in, const int* in_sizes, int n_in,
                              void* d_out, int out_size, void* d_ws, size_t ws_size,
                              hipStream_t stream) {
    const float* srcF  = (const float*)d_in[0];
    const float* edgeF = (const float*)d_in[1];
    const float* W1    = (const float*)d_in[2];
    const float* B1    = (const float*)d_in[3];
    const float* GAMMA = (const float*)d_in[4];
    const float* BETA  = (const float*)d_in[5];
    const float* W2    = (const float*)d_in[6];
    const float* B2    = (const float*)d_in[7];
    float* out = (float*)d_out;

    const int E = in_sizes[0] / 128;
    const int nTiles = (E + 15) / 16;

    int blocks = (nTiles + WAVES_PER_BLOCK - 1) / WAVES_PER_BLOCK;
    if (blocks > 1024) blocks = 1024;   // persistent waves amortize weight staging
    if (blocks < 1) blocks = 1;

    edge_mlp_wmma_kernel<<<blocks, TPB, 0, stream>>>(
        srcF, edgeF, W1, B1, GAMMA, BETA, W2, B2, out, E, nTiles);
}
